// GraphDownBlock_80676665688604
// MI455X (gfx1250) — matrix operand order repaired and checked by reading, not compiled, not run on hardware
//
#include <hip/hip_runtime.h>

#define NN 50000
#define EE 800000
#define DF 64
#define HH 4
#define CC 16

typedef _Float16 h8_t  __attribute__((ext_vector_type(8)));
typedef _Float16 v16h  __attribute__((ext_vector_type(16)));
typedef float    v8f   __attribute__((ext_vector_type(8)));

__device__ __forceinline__ float silu_f(float v) { return v / (1.0f + __expf(-v)); }
__device__ __forceinline__ float lrelu_f(float v) { return v > 0.0f ? v : 0.2f * v; }

// Monotonic float<->uint encoding for atomicMax-based segment max.
__device__ __forceinline__ unsigned fenc(float f) {
    unsigned u = __float_as_uint(f);
    return (u & 0x80000000u) ? ~u : (u | 0x80000000u);
}
__device__ __forceinline__ float fdec(unsigned u) {
    return __uint_as_float((u & 0x80000000u) ? (u & 0x7fffffffu) : ~u);
}

// ---------------- utility kernels ----------------
__global__ void k_zero_f32(float* p, int n) {
    int t = blockIdx.x * blockDim.x + threadIdx.x;
    if (t < n) p[t] = 0.0f;
}
__global__ void k_zero_u32(unsigned* p, int n) {
    int t = blockIdx.x * blockDim.x + threadIdx.x;
    if (t < n) p[t] = 0u;
}
__global__ void k_f32_to_f16pair(const float* __restrict__ src,
                                 _Float16* __restrict__ hi, _Float16* __restrict__ lo, int n) {
    int t = blockIdx.x * blockDim.x + threadIdx.x;
    if (t < n) {
        float v = src[t];
        _Float16 h = (_Float16)v;
        hi[t] = h;
        lo[t] = (_Float16)(v - (float)h);
    }
}
__global__ void k_sum_reduce(const float* __restrict__ x, int n, float* out) {
    __shared__ float sm[256];
    int tid = threadIdx.x;
    int gid = blockIdx.x * 256 + tid;
    sm[tid] = (gid < n) ? x[gid] : 0.0f;
    __syncthreads();
    for (int s = 128; s > 0; s >>= 1) {
        if (tid < s) sm[tid] += sm[tid + s];
        __syncthreads();
    }
    if (tid == 0) atomicAdd(out, sm[0]);
}

// ---------------- degree / norm ----------------
__global__ void k_deg(const int* __restrict__ ei, const float* __restrict__ ew, float* deg) {
    int e = blockIdx.x * blockDim.x + threadIdx.x;
    if (e < EE) atomicAdd(&deg[ei[EE + e]], ew[e]);
}
__global__ void k_dinv(float* d) {   // in-place: deg -> rsqrt(deg + self_loop)
    int i = blockIdx.x * blockDim.x + threadIdx.x;
    if (i < NN) d[i] = rsqrtf(d[i] + 1.0f);
}
__global__ void k_norm(const int* __restrict__ ei, const float* __restrict__ ew,
                       const float* __restrict__ dinv, float* __restrict__ nrm) {
    int e = blockIdx.x * blockDim.x + threadIdx.x;
    if (e < EE) nrm[e] = dinv[ei[e]] * ew[e] * dinv[ei[EE + e]];
}

// ---------------- WMMA GEMM: Y[n x 64] = X[n x 64] @ W[64 x 64] (+bias) ----------------
// fp32 emulated via f16 hi/lo split: X*W ~= Xh*Wh + Xl*Wh + Xh*Wl  (f32 accumulate)
__global__ __launch_bounds__(128)
void k_gemm64(const _Float16* __restrict__ Ahi, const _Float16* __restrict__ Alo,
              const _Float16* __restrict__ Bhi, const _Float16* __restrict__ Blo,
              const float* __restrict__ bias, float* __restrict__ Y) {
    const int wave = threadIdx.x >> 5;       // 0..3 -> 16-col tile
    const int lane = threadIdx.x & 31;
    const int hv   = lane >> 4;              // lane half (K-group select)
    const int m    = lane & 15;              // A row within tile
    const int row  = blockIdx.x * 16 + m;    // NN is a multiple of 16
    const int woff = wave * 16;

    v8f acc = {};
#pragma unroll
    for (int ks = 0; ks < 2; ++ks) {
        const int kofs = ks * 32;
        // A layout (16-bit 16x32): lane m, VGPR0-3 -> K = 2v+j + 8*half, VGPR4-7 -> +16
        const _Float16* arh = Ahi + (size_t)row * DF + kofs + 8 * hv;
        const _Float16* arl = Alo + (size_t)row * DF + kofs + 8 * hv;
        h8_t a0h = *(const h8_t*)(arh);
        h8_t a1h = *(const h8_t*)(arh + 16);
        h8_t a0l = *(const h8_t*)(arl);
        h8_t a1l = *(const h8_t*)(arl + 16);
        v16h ahi = __builtin_shufflevector(a0h, a1h, 0,1,2,3,4,5,6,7,8,9,10,11,12,13,14,15);
        v16h alo = __builtin_shufflevector(a0l, a1l, 0,1,2,3,4,5,6,7,8,9,10,11,12,13,14,15);
        // B layout (16-bit 32x16): lane n = lane&15, elements -> K same as A (8*half + s, +16 for s>=8)
        v16h bhi, blo;
#pragma unroll
        for (int s = 0; s < 16; ++s) {
            const int krow = kofs + 8 * hv + (s < 8 ? s : s + 8);
            bhi[s] = Bhi[(size_t)krow * DF + woff + m];
            blo[s] = Blo[(size_t)krow * DF + woff + m];
        }
        acc = __builtin_amdgcn_wmma_f32_16x16x32_f16(false, ahi, false, bhi, (short)0, acc, false, false);
        acc = __builtin_amdgcn_wmma_f32_16x16x32_f16(false, alo, false, bhi, (short)0, acc, false, false);
        acc = __builtin_amdgcn_wmma_f32_16x16x32_f16(false, ahi, false, blo, (short)0, acc, false, false);
    }
    // D layout (32-bit 16x16): lane n = lane&15, VGPR v -> M = v + 8*(lane>>4)
    const int ncol = lane & 15;
    const float bval = bias ? bias[woff + ncol] : 0.0f;
#pragma unroll
    for (int v = 0; v < 8; ++v) {
        int orow = blockIdx.x * 16 + v + 8 * hv;
        Y[(size_t)orow * DF + woff + ncol] = acc[v] + bval;
    }
}

// ---------------- GCN edge aggregation ----------------
__global__ void k_gcn_edge(const int* __restrict__ ei, const float* __restrict__ nrm,
                           const float* __restrict__ h, float* __restrict__ agg) {
    long long t = (long long)blockIdx.x * blockDim.x + threadIdx.x;
    if (t >= (long long)EE * DF) return;
    int e = (int)(t >> 6), f = (int)(t & 63);
    atomicAdd(&agg[(size_t)ei[EE + e] * DF + f], h[(size_t)ei[e] * DF + f] * nrm[e]);
}

// GCN self-loop + bias + SiLU + residual + LayerNorm (one wave per node)
__global__ __launch_bounds__(256)
void k_gcn_post_ln(const float* __restrict__ agg, const float* __restrict__ h,
                   const float* __restrict__ dinv, const float* __restrict__ gb,
                   const float* __restrict__ res, const float* __restrict__ lg,
                   const float* __restrict__ lb, float* __restrict__ xm) {
    int wave = threadIdx.x >> 5, lane = threadIdx.x & 31;
    int i = blockIdx.x * 8 + wave;
    if (i >= NN) return;
    float dv = dinv[i], d2 = dv * dv;
    float t0, t1;
    {
        size_t idx = (size_t)i * DF + lane;
        float v = agg[idx] + h[idx] * d2 + gb[lane];
        t0 = silu_f(v) + res[idx];
        idx += 32;
        v = agg[idx] + h[idx] * d2 + gb[lane + 32];
        t1 = silu_f(v) + res[idx];
    }
    float sum = t0 + t1, sq = t0 * t0 + t1 * t1;
    for (int off = 16; off > 0; off >>= 1) {
        sum += __shfl_xor(sum, off, 32);
        sq  += __shfl_xor(sq, off, 32);
    }
    float mu = sum * (1.0f / 64.0f);
    float var = sq * (1.0f / 64.0f) - mu * mu;
    float rs = rsqrtf(var + 1e-5f);
    size_t idx = (size_t)i * DF + lane;
    xm[idx]      = (t0 - mu) * rs * lg[lane]      + lb[lane];
    xm[idx + 32] = (t1 - mu) * rs * lg[lane + 32] + lb[lane + 32];
}

// ---------------- GAT ----------------
__global__ void k_edot(const float* __restrict__ We, const float* __restrict__ ae, float* edot) {
    int t = threadIdx.x;
    if (t < HH) {
        float s = 0.0f;
        for (int c = 0; c < CC; ++c) s += We[t * CC + c] * ae[t * CC + c];
        edot[t] = s;
    }
}
// per (node, head): a_src/a_dst dots + self-loop alpha + seed segment max
__global__ void k_gat_node(const float* __restrict__ xl, const float* __restrict__ asv,
                           const float* __restrict__ adv, const float* __restrict__ edot,
                           const float* __restrict__ easum, float* __restrict__ a_s,
                           float* __restrict__ a_d, float* __restrict__ selfb,
                           unsigned* __restrict__ mbuf) {
    int t = blockIdx.x * blockDim.x + threadIdx.x;
    if (t >= NN * HH) return;
    int i = t >> 2, hh = t & 3;
    float as = 0.0f, ad = 0.0f;
    for (int c = 0; c < CC; ++c) {
        float xv = xl[(size_t)i * DF + hh * CC + c];
        as += xv * asv[hh * CC + c];
        ad += xv * adv[hh * CC + c];
    }
    a_s[t] = as; a_d[t] = ad;
    float mea = easum[0] * (1.0f / (float)EE);
    float al = lrelu_f(as + ad + mea * edot[hh]);
    selfb[t] = al;
    atomicMax(&mbuf[t], fenc(al));
}
__global__ void k_gat_edge_max(const int* __restrict__ ei, const float* __restrict__ ea,
                               const float* __restrict__ edot, const float* __restrict__ a_s,
                               const float* __restrict__ a_d, float* __restrict__ alpha,
                               unsigned* __restrict__ mbuf) {
    int t = blockIdx.x * blockDim.x + threadIdx.x;
    if (t >= EE * HH) return;
    int e = t >> 2, hh = t & 3;
    int s = ei[e], d = ei[EE + e];
    float al = lrelu_f(a_s[s * HH + hh] + a_d[d * HH + hh] + ea[e] * edot[hh]);
    alpha[t] = al;
    atomicMax(&mbuf[d * HH + hh], fenc(al));
}
__global__ void k_gat_edge_exp(const int* __restrict__ ei, float* __restrict__ alpha,
                               const unsigned* __restrict__ mbuf, float* __restrict__ sbuf) {
    int t = blockIdx.x * blockDim.x + threadIdx.x;
    if (t >= EE * HH) return;
    int e = t >> 2, hh = t & 3;
    int d = ei[EE + e];
    float ex = __expf(alpha[t] - fdec(mbuf[d * HH + hh]));
    alpha[t] = ex;
    atomicAdd(&sbuf[d * HH + hh], ex);
}
__global__ void k_gat_self_exp(float* __restrict__ selfb, const unsigned* __restrict__ mbuf,
                               float* __restrict__ sbuf) {
    int t = blockIdx.x * blockDim.x + threadIdx.x;
    if (t >= NN * HH) return;
    float ex = __expf(selfb[t] - fdec(mbuf[t]));
    selfb[t] = ex;
    atomicAdd(&sbuf[t], ex);
}
__global__ void k_gat_edge_aggr(const int* __restrict__ ei, const float* __restrict__ alpha,
                                const float* __restrict__ sbuf, const float* __restrict__ xl,
                                float* __restrict__ out) {
    long long t = (long long)blockIdx.x * blockDim.x + threadIdx.x;
    if (t >= (long long)EE * DF) return;
    int e = (int)(t >> 6), f = (int)(t & 63), hh = f >> 4;
    int s = ei[e], d = ei[EE + e];
    float coef = alpha[e * HH + hh] / (sbuf[d * HH + hh] + 1e-16f);
    atomicAdd(&out[(size_t)d * DF + f], xl[(size_t)s * DF + f] * coef);
}
// self contribution + bias + SiLU + residual
__global__ void k_gat_post(const float* __restrict__ agg, const float* __restrict__ xl,
                           const float* __restrict__ selfb, const float* __restrict__ sbuf,
                           const float* __restrict__ bg, const float* __restrict__ xm,
                           float* __restrict__ xout) {
    int t = blockIdx.x * blockDim.x + threadIdx.x;
    if (t >= NN * DF) return;
    int i = t >> 6, f = t & 63, hh = f >> 4;
    float coef = selfb[i * HH + hh] / (sbuf[i * HH + hh] + 1e-16f);
    float v = agg[t] + xl[t] * coef + bg[f];
    xout[t] = silu_f(v) + xm[t];
}

// ---------------- host orchestration ----------------
static inline unsigned cdiv(long long a, long long b) { return (unsigned)((a + b - 1) / b); }

extern "C" void kernel_launch(void* const* d_in, const int* in_sizes, int n_in,
                              void* d_out, int out_size, void* d_ws, size_t ws_size,
                              hipStream_t stream) {
    (void)in_sizes; (void)n_in; (void)out_size; (void)ws_size;
    const float* x_in = (const float*)d_in[0];
    const int*   ei   = (const int*)d_in[1];
    const float* ew   = (const float*)d_in[2];
    const float* ea   = (const float*)d_in[3];

    // workspace carve-up
    size_t off = 0;
    auto carve = [&](size_t bytes) {
        void* p = (char*)d_ws + off;
        off = (off + bytes + 255) & ~(size_t)255;
        return p;
    };
    _Float16* xhi = (_Float16*)carve((size_t)NN * DF * 2);
    _Float16* xlo = (_Float16*)carve((size_t)NN * DF * 2);
    _Float16* whi = (_Float16*)carve((size_t)6 * DF * DF * 2);
    _Float16* wlo = (_Float16*)carve((size_t)6 * DF * DF * 2);
    float* hbuf   = (float*)carve((size_t)NN * DF * 4);   // also reused as xl
    float* resbuf = (float*)carve((size_t)NN * DF * 4);   // also reused as alpha (E*4 floats)
    float* aggbuf = (float*)carve((size_t)NN * DF * 4);
    float* xmbuf  = (float*)carve((size_t)NN * DF * 4);
    float* xabuf  = (float*)carve((size_t)NN * DF * 4);
    float* dinvb  = (float*)carve((size_t)NN * 4);
    float* nrmb   = (float*)carve((size_t)EE * 4);
    float* asb    = (float*)carve((size_t)NN * HH * 4);
    float* adb    = (float*)carve((size_t)NN * HH * 4);
    float* selfb  = (float*)carve((size_t)NN * HH * 4);
    unsigned* mb  = (unsigned*)carve((size_t)NN * HH * 4);
    float* sb     = (float*)carve((size_t)NN * HH * 4);
    float* scal   = (float*)carve(256);                   // [0]=sum(ea), [8..11]=edot
    float* xlbuf  = hbuf;
    float* alphab = resbuf;
    float* edotp  = scal + 8;

    // degree / symmetric norm (shared across layers)
    k_zero_f32<<<cdiv(NN, 256), 256, 0, stream>>>(dinvb, NN);
    k_deg<<<cdiv(EE, 256), 256, 0, stream>>>(ei, ew, dinvb);
    k_dinv<<<cdiv(NN, 256), 256, 0, stream>>>(dinvb);
    k_norm<<<cdiv(EE, 256), 256, 0, stream>>>(ei, ew, dinvb, nrmb);
    // mean(edge_attr)
    k_zero_f32<<<1, 256, 0, stream>>>(scal, 1);
    k_sum_reduce<<<cdiv(EE, 256), 256, 0, stream>>>(ea, EE, scal);
    // convert all 6 weight matrices to f16 hi/lo pairs
    const int widx[6] = {4, 6, 10, 16, 18, 22}; // gcn_w0, res_w0, gat_w0, gcn_w1, res_w1, gat_w1
    for (int j = 0; j < 6; ++j)
        k_f32_to_f16pair<<<cdiv(DF * DF, 256), 256, 0, stream>>>(
            (const float*)d_in[widx[j]], whi + j * DF * DF, wlo + j * DF * DF, DF * DF);

    const float* xcur = x_in;
    float* xouts[2] = { xabuf, (float*)d_out };
    for (int l = 0; l < 2; ++l) {
        const int b = 4 + l * 12;
        const float* gb   = (const float*)d_in[b + 1];
        const float* rb   = (const float*)d_in[b + 3];
        const float* lg   = (const float*)d_in[b + 4];
        const float* lb   = (const float*)d_in[b + 5];
        const float* bg   = (const float*)d_in[b + 7];
        const float* asv  = (const float*)d_in[b + 8];
        const float* adv  = (const float*)d_in[b + 9];
        const float* aev  = (const float*)d_in[b + 10];
        const float* Wev  = (const float*)d_in[b + 11];
        const _Float16* gwh = whi + (l * 3 + 0) * DF * DF, * gwl = wlo + (l * 3 + 0) * DF * DF;
        const _Float16* rwh = whi + (l * 3 + 1) * DF * DF, * rwl = wlo + (l * 3 + 1) * DF * DF;
        const _Float16* awh = whi + (l * 3 + 2) * DF * DF, * awl = wlo + (l * 3 + 2) * DF * DF;

        // GEMMs: residual (with bias) and GCN transform
        k_f32_to_f16pair<<<cdiv((size_t)NN * DF, 256), 256, 0, stream>>>(xcur, xhi, xlo, NN * DF);
        k_gemm64<<<NN / 16, 128, 0, stream>>>(xhi, xlo, rwh, rwl, rb, resbuf);
        k_gemm64<<<NN / 16, 128, 0, stream>>>(xhi, xlo, gwh, gwl, nullptr, hbuf);
        // GCN aggregation + fused epilogue/LN
        k_zero_f32<<<cdiv((size_t)NN * DF, 256), 256, 0, stream>>>(aggbuf, NN * DF);
        k_gcn_edge<<<cdiv((long long)EE * DF, 256), 256, 0, stream>>>(ei, nrmb, hbuf, aggbuf);
        k_gcn_post_ln<<<cdiv(NN, 8), 256, 0, stream>>>(aggbuf, hbuf, dinvb, gb, resbuf, lg, lb, xmbuf);

        // GAT
        k_edot<<<1, 64, 0, stream>>>(Wev, aev, edotp);
        k_f32_to_f16pair<<<cdiv((size_t)NN * DF, 256), 256, 0, stream>>>(xmbuf, xhi, xlo, NN * DF);
        k_gemm64<<<NN / 16, 128, 0, stream>>>(xhi, xlo, awh, awl, nullptr, xlbuf);
        k_zero_u32<<<cdiv(NN * HH, 256), 256, 0, stream>>>(mb, NN * HH);
        k_zero_f32<<<cdiv(NN * HH, 256), 256, 0, stream>>>(sb, NN * HH);
        k_gat_node<<<cdiv(NN * HH, 256), 256, 0, stream>>>(xlbuf, asv, adv, edotp, scal, asb, adb, selfb, mb);
        k_gat_edge_max<<<cdiv(EE * HH, 256), 256, 0, stream>>>(ei, ea, edotp, asb, adb, alphab, mb);
        k_gat_edge_exp<<<cdiv(EE * HH, 256), 256, 0, stream>>>(ei, alphab, mb, sb);
        k_gat_self_exp<<<cdiv(NN * HH, 256), 256, 0, stream>>>(selfb, mb, sb);
        k_zero_f32<<<cdiv((size_t)NN * DF, 256), 256, 0, stream>>>(aggbuf, NN * DF);
        k_gat_edge_aggr<<<cdiv((long long)EE * DF, 256), 256, 0, stream>>>(ei, alphab, sb, xlbuf, aggbuf);
        k_gat_post<<<cdiv((size_t)NN * DF, 256), 256, 0, stream>>>(aggbuf, xlbuf, selfb, sb, bg, xmbuf, xouts[l]);

        xcur = xouts[l];
    }
}
